// MyModel_71914932404692
// MI455X (gfx1250) — matrix operand order repaired and checked by reading, not compiled, not run on hardware
//
#include <hip/hip_runtime.h>

// ---------------- problem constants ----------------
#define TT 30
#define BB 512
#define HH 2048
#define GG 300

// ---------------- WMMA types ----------------
typedef __attribute__((ext_vector_type(16))) _Float16 v16h;
typedef __attribute__((ext_vector_type(8)))  _Float16 v8h;
typedef __attribute__((ext_vector_type(4)))  _Float16 v4h;
typedef __attribute__((ext_vector_type(8)))  float    v8f;

// ---------------- tiled NT GEMM: C[M,N] = A[M,K] * W[N,K]^T (+ bias[N]) ----
// A row-major (lda), W row-major (ldw), C row-major (ldc), f32 in/out,
// f16 multiply with f32 accumulate via v_wmma_f32_16x16x32_f16.
// Software-pipelined: double-buffered LDS, global->reg preload of tile k+1
// overlaps WMMA of tile k. One barrier per K-tile.
// Requires: M % 128 == 0 (true for all call sites here). N, K arbitrary.
#define BM 128
#define BN 64
#define BK 32
#define SA_LD 40   // _Float16 stride for sA rows (80B: 16B-aligned frags)
#define SB_LD 48   // _Float16 stride for sB rows (96B: 32B-aligned frags)

__global__ __launch_bounds__(256)
void gemm_nt_f16(const float* __restrict__ A, int lda,
                 const float* __restrict__ W, int ldw,
                 const float* __restrict__ bias,
                 float* __restrict__ C, int ldc,
                 int M, int N, int K)
{
    __shared__ __attribute__((aligned(32))) _Float16 sA[2][BM * SA_LD];
    __shared__ __attribute__((aligned(32))) _Float16 sB[2][BN * SB_LD];

    const int tid  = threadIdx.x;
    const int wv   = tid >> 5;          // wave 0..7 -> M sub-strip
    const int lane = tid & 31;
    const int kh   = lane >> 4;         // lane half (0/1)
    const int l16  = lane & 15;

    const long mBlock = (long)blockIdx.y * BM;
    const long nBlock = (long)blockIdx.x * BN;

    // staging coordinates: thread -> (row srow + 32*i, cols scol..scol+3)
    const int srow = tid >> 3;          // 0..31
    const int scol = (tid & 7) * 4;     // 0,4,...,28

    float4 aReg[4];                     // A rows srow, srow+32, +64, +96
    float4 bReg[2];                     // W rows srow, srow+32

    v8f acc[4] = {};

    const int nTiles = (K + BK - 1) / BK;

    // ---- global f32 -> registers (uniform fast/tail branch per tile) ----
    auto load_tile = [&](int kb) {
        if (kb + BK <= K) {             // uniform: whole tile in-bounds in K
#pragma unroll
            for (int i = 0; i < 4; ++i)
                aReg[i] = *(const float4*)(A + (mBlock + srow + 32 * i) * (long)lda + kb + scol);
#pragma unroll
            for (int i = 0; i < 2; ++i) {
                const long n = nBlock + srow + 32 * i;
                bReg[i] = (n < N) ? *(const float4*)(W + n * (long)ldw + kb + scol)
                                  : make_float4(0.f, 0.f, 0.f, 0.f);
            }
        } else {                        // K tail (only last tile of K=300 GEMMs)
#pragma unroll
            for (int i = 0; i < 4; ++i) {
                const float* src = A + (mBlock + srow + 32 * i) * (long)lda + kb + scol;
                float v[4];
#pragma unroll
                for (int j = 0; j < 4; ++j)
                    v[j] = (kb + scol + j < K) ? src[j] : 0.f;
                aReg[i] = make_float4(v[0], v[1], v[2], v[3]);
            }
#pragma unroll
            for (int i = 0; i < 2; ++i) {
                const long n = nBlock + srow + 32 * i;
                float v[4] = {0.f, 0.f, 0.f, 0.f};
                if (n < N) {
                    const float* src = W + n * (long)ldw + kb + scol;
#pragma unroll
                    for (int j = 0; j < 4; ++j)
                        if (kb + scol + j < K) v[j] = src[j];
                }
                bReg[i] = make_float4(v[0], v[1], v[2], v[3]);
            }
        }
    };

    // ---- registers -> f16 LDS (packed 4xf16 = ds_store_b64) ----
    auto store_tile = [&](int buf) {
#pragma unroll
        for (int i = 0; i < 4; ++i) {
            v4h p = { (_Float16)aReg[i].x, (_Float16)aReg[i].y,
                      (_Float16)aReg[i].z, (_Float16)aReg[i].w };
            *(v4h*)&sA[buf][(srow + 32 * i) * SA_LD + scol] = p;
        }
#pragma unroll
        for (int i = 0; i < 2; ++i) {
            v4h p = { (_Float16)bReg[i].x, (_Float16)bReg[i].y,
                      (_Float16)bReg[i].z, (_Float16)bReg[i].w };
            *(v4h*)&sB[buf][(srow + 32 * i) * SB_LD + scol] = p;
        }
    };

    load_tile(0);

    for (int it = 0; it < nTiles; ++it) {
        const int buf = it & 1;
        store_tile(buf);
        __syncthreads();                       // one barrier per tile

        if (it + 1 < nTiles) load_tile((it + 1) * BK);   // overlap with WMMA

        // A fragment (16x32): lane l16 = row; elems 0-15 = K kh*16+0..15
        // contiguous (matches B)
        const _Float16* ar = &sA[buf][(wv * 16 + l16) * SA_LD];
        v8h alo = *(const v8h*)(ar + kh * 16);
        v8h ahi = *(const v8h*)(ar + kh * 16 + 8);
        v16h afrag = __builtin_shufflevector(alo, ahi,
                        0,1,2,3,4,5,6,7,8,9,10,11,12,13,14,15);

        // B fragments (32x16): lane l16 = column n (a W row);
        // elems 0-15 = K kh*16+0..15 contiguous.  Load all, then 4 WMMAs.
        v16h bf[4];
#pragma unroll
        for (int nt = 0; nt < 4; ++nt)
            bf[nt] = *(const v16h*)&sB[buf][(nt * 16 + l16) * SB_LD + kh * 16];
#pragma unroll
        for (int nt = 0; nt < 4; ++nt)
            acc[nt] = __builtin_amdgcn_wmma_f32_16x16x32_f16(
                          false, afrag, false, bf[nt],
                          (short)0, acc[nt], false, false);
    }

    // ---- store: element r of v8f is M = r + 8*kh, N = l16 ----
    const long mW = mBlock + wv * 16 + kh * 8;
#pragma unroll
    for (int nt = 0; nt < 4; ++nt) {
        const long n = nBlock + nt * 16 + l16;
        if (n < N) {
            const float bv = bias ? bias[n] : 0.f;
#pragma unroll
            for (int r = 0; r < 8; ++r)
                C[(mW + r) * (long)ldc + n] = acc[nt][r] + bv;
        }
    }
}

// ---------------- GRU gate fusion ----------------
// xp: (B,3G) slice for step t; gh: (B,3G) = h@w_hh^T + b_hh;
// h updated in place; h_out = h_de[t] slice of d_out.
__global__ void gru_gate(const float* __restrict__ xp,
                         const float* __restrict__ gh,
                         float* __restrict__ h,
                         float* __restrict__ h_out)
{
    int i = blockIdx.x * blockDim.x + threadIdx.x;
    if (i >= BB * GG) return;
    int b = i / GG, g = i - b * GG;
    const float* xr = xp + (long)b * (3 * GG);
    const float* gr = gh + (long)b * (3 * GG);
    float r = 1.f / (1.f + expf(-(xr[g]          + gr[g])));
    float z = 1.f / (1.f + expf(-(xr[GG + g]     + gr[GG + g])));
    float n = tanhf(xr[2 * GG + g] + r * gr[2 * GG + g]);
    float hn = (1.f - z) * n + z * h[i];
    h[i] = hn;
    h_out[i] = hn;
}

// ---------------- attention: scores -> softmax -> context ----------------
// he: (B,G) = h_new @ w_sc_h^T ; yW: (T,B,G) precomputed (incl. sc_b)
// scores[b,t'] = sum_g agg[g]*tanh(he[b,g]+yW[t',b,g]); ctx = softmax @ y.
__global__ __launch_bounds__(256)
void attention(const float* __restrict__ he,
               const float* __restrict__ yW,
               const float* __restrict__ agg,
               const float* __restrict__ y,
               float* __restrict__ ctx)
{
    const int b = blockIdx.x;
    __shared__ float s_he[GG];
    __shared__ float s_agg[GG];
    __shared__ float s_scores[32];
    __shared__ float s_att[32];

    for (int g = threadIdx.x; g < GG; g += 256) {
        s_he[g]  = he[(long)b * GG + g];
        s_agg[g] = agg[g];
    }
    __syncthreads();

    const int wv = threadIdx.x >> 5, lane = threadIdx.x & 31;
    for (int t = wv; t < TT; t += 8) {
        const float* yWp = yW + ((long)t * BB + b) * GG;
        float sum = 0.f;
        for (int g = lane; g < GG; g += 32)
            sum += s_agg[g] * tanhf(s_he[g] + yWp[g]);
#pragma unroll
        for (int off = 16; off > 0; off >>= 1)
            sum += __shfl_xor(sum, off, 32);
        if (lane == 0) s_scores[t] = sum;
    }
    __syncthreads();

    if (threadIdx.x < 32) {
        float v = (lane < TT) ? s_scores[lane] : -1e30f;
        float m = v;
#pragma unroll
        for (int off = 16; off > 0; off >>= 1)
            m = fmaxf(m, __shfl_xor(m, off, 32));
        float e = (lane < TT) ? expf(v - m) : 0.f;
        float s = e;
#pragma unroll
        for (int off = 16; off > 0; off >>= 1)
            s += __shfl_xor(s, off, 32);
        if (lane < TT) s_att[lane] = e / s;
    }
    __syncthreads();

    for (int hh = threadIdx.x; hh < HH; hh += 256) {
        float acc = 0.f;
#pragma unroll 5
        for (int t = 0; t < TT; ++t)
            acc += s_att[t] * y[((long)t * BB + b) * HH + hh];
        ctx[(long)b * HH + hh] = acc;
    }
}

// ---------------- host-side launch ----------------
extern "C" void kernel_launch(void* const* d_in, const int* in_sizes, int n_in,
                              void* d_out, int out_size, void* d_ws, size_t ws_size,
                              hipStream_t stream)
{
    const float* y       = (const float*)d_in[0];   // (T,B,H)
    const float* context = (const float*)d_in[1];   // (B,H)
    const float* hm_w    = (const float*)d_in[2];   // (G,H)
    const float* hm_b    = (const float*)d_in[3];   // (G)
    const float* sc_w    = (const float*)d_in[4];   // (G, G+H)
    const float* sc_b    = (const float*)d_in[5];   // (G)
    const float* agg     = (const float*)d_in[6];   // (G)
    const float* w_ih    = (const float*)d_in[7];   // (3G,H)
    const float* w_hh    = (const float*)d_in[8];   // (3G,G)
    const float* b_ih    = (const float*)d_in[9];   // (3G)
    const float* b_hh    = (const float*)d_in[10];  // (3G)

    float* out = (float*)d_out;                 // [T*B*G h_de | T*B*H ctx]
    float* ws  = (float*)d_ws;

    float* x_proj = ws;                                   // (T*B, 3G)
    float* yW     = x_proj + (long)TT * BB * 3 * GG;      // (T*B, G)
    float* h      = yW     + (long)TT * BB * GG;          // (B, G)
    float* gh     = h      + (long)BB * GG;               // (B, 3G)
    float* he     = gh     + (long)BB * 3 * GG;           // (B, G)

    float* h_de = out;
    float* ctxv = out + (long)TT * BB * GG;

    const dim3 blk(256);
    // h0 = context @ hm_w^T + hm_b       : M=512, N=300, K=2048
    gemm_nt_f16<<<dim3(5, 4),   blk, 0, stream>>>(context, HH, hm_w, HH, hm_b, h,  GG,     BB,      GG,     HH);
    // yW = y @ w_sc_y^T + sc_b           : M=15360, N=300, K=2048 (W = sc_w[:,G:], row stride G+H)
    gemm_nt_f16<<<dim3(5, 120), blk, 0, stream>>>(y, HH, sc_w + GG, GG + HH, sc_b, yW, GG, TT * BB, GG,     HH);
    // x_proj = y @ w_ih^T + b_ih         : M=15360, N=900, K=2048
    gemm_nt_f16<<<dim3(15, 120), blk, 0, stream>>>(y, HH, w_ih, HH, b_ih, x_proj, 3 * GG, TT * BB, 3 * GG, HH);

    for (int t = 0; t < TT; ++t) {
        // gh = h @ w_hh^T + b_hh         : M=512, N=900, K=300
        gemm_nt_f16<<<dim3(15, 4), blk, 0, stream>>>(h, GG, w_hh, GG, b_hh, gh, 3 * GG, BB, 3 * GG, GG);
        // gates + state update + h_de[t]
        gru_gate<<<(BB * GG + 255) / 256, 256, 0, stream>>>(
            x_proj + (long)t * BB * 3 * GG, gh, h, h_de + (long)t * BB * GG);
        // he = h_new @ w_sc_h^T          : M=512, N=300, K=300 (W = sc_w[:,:G], row stride G+H)
        gemm_nt_f16<<<dim3(5, 4), blk, 0, stream>>>(h, GG, sc_w, GG + HH, nullptr, he, GG, BB, GG, GG);
        // attention -> ctx_vec[t]
        attention<<<BB, 256, 0, stream>>>(he, yW, agg, y, ctxv + (long)t * BB * HH);
    }
}
